// ODE_VAE_Decoder_20942260535805
// MI455X (gfx1250) — matrix operand order repaired and checked by reading, not compiled, not run on hardware
//
#include <hip/hip_runtime.h>
#include <hip/hip_bf16.h>

// ---------------------------------------------------------------------------
// ODE-VAE decoder for MI455X (gfx1250): bf16 WMMA GEMMs, f32 state/accum.
// Round 4: per-stage fragment preload (clause ds-loads, partial dscnt waits,
// 8 back-to-back WMMAs); template BN so GEMM2/projection get 2x the blocks.
// ---------------------------------------------------------------------------

typedef __attribute__((ext_vector_type(16))) __bf16 v16bf;
typedef __attribute__((ext_vector_type(8)))  float  v8f;
typedef __attribute__((ext_vector_type(4)))  float  v4f;
typedef __attribute__((ext_vector_type(4)))  unsigned v4u;

static constexpr int kB  = 512;
static constexpr int kH  = 1024;
static constexpr int kOH = 4096;
static constexpr int kC  = 512;
static constexpr int kT  = 40;

// GEMM tiling
static constexpr int BM  = 64;     // block rows
static constexpr int KC  = 64;     // K-chunk per LDS stage (2 WMMA k-steps)
static constexpr int LDP = KC + 8; // padded LDS row stride (144B: bank-safe)

__device__ __forceinline__ __bf16 to_bf16(float x) { return (__bf16)x; }

// Hardware tanh (TRANS32 op on CDNA5). ISA 7.4: one independent op / V_NOP
// required after a TRANS op before its result is consumed.
__device__ __forceinline__ float fast_tanh(float x) {
    float r;
    asm volatile("v_tanh_f32 %0, %1\n\tv_nop" : "=v"(r) : "v"(x));
    return r;
}

// Async 16B/lane copy global -> LDS (VDST = LDS byte offset, VADDR = global).
// Tracked by ASYNCcnt; completion drained with s_wait_asynccnt.
__device__ __forceinline__ void async_copy16(unsigned lds_off, const void* gaddr) {
    asm volatile("global_load_async_to_lds_b128 %0, %1, off"
                 :: "v"(lds_off), "v"(gaddr) : "memory");
}

// ---------------------------------------------------------------------------
// WMMA fragments (ISA 7.12.2, wave32)
// ---------------------------------------------------------------------------
union Frag {
    v16bf v;
    v4u   q[2];
};

__device__ __forceinline__ v8f wmma_bf16(const Frag& a, const Frag& b, v8f c) {
    return __builtin_amdgcn_wmma_f32_16x16x32_bf16(false, a.v, false, b.v,
                                                   (short)0, c, false, false);
}

// ---------------------------------------------------------------------------
// GEMM: D[M,N] = A[M,K] (bf16 row-major) x Bt[N,K] (bf16 N-major) + bias[N]
// Block = 256 threads (8 waves): 4 row groups x 2 col groups.
// Block tile BM x BN_; wave tile 16 x (BN_/2); 2-stage async->LDS pipeline.
// EPI 0: tanh -> bf16 activations
// EPI 1: f32 + bias (projection logits)
// EPI 2: RK4 3/8-rule combine fused
// ---------------------------------------------------------------------------
template <int EPI, int BN_>
__global__ __launch_bounds__(256)
void wmma_gemm_kernel(const __bf16* __restrict__ A,
                      const __bf16* __restrict__ Bt,
                      const float*  __restrict__ bias,
                      float*        __restrict__ Df,
                      __bf16*       __restrict__ Dh,
                      int M, int N, int K,
                      int mode, const float* __restrict__ ts, int tidx,
                      float* __restrict__ y,
                      float* __restrict__ kv1, float* __restrict__ kv2,
                      float* __restrict__ kv3,
                      __bf16* __restrict__ ubf, __bf16* __restrict__ ybf)
{
    constexpr int NJ     = BN_ / 32;        // 16x16 tiles per wave (4 or 2)
    constexpr int NBCHK  = BN_ / 32;        // B copy chunks per thread
    constexpr int NASYNC = 2 + NBCHK;       // async copies per thread per stage

    __shared__ alignas(16) __bf16 As[2][BM][LDP];
    __shared__ alignas(16) __bf16 Bs[2][BN_][LDP];

    const int tid  = threadIdx.x;
    const int w    = tid >> 5;
    const int wr   = w >> 1;                // 0..3 row group (16 rows each)
    const int wc   = w & 1;                 // 0..1 col group (BN_/2 cols each)
    const int lane = tid & 31;
    const int m0   = blockIdx.y * BM;
    const int n0   = blockIdx.x * BN_;

    const __bf16* Ab = A  + (size_t)m0 * K;
    const __bf16* Bb = Bt + (size_t)n0 * K;

    // Stage copy: A BMx64 in 2 chunks/thread, B BN_x64 in NBCHK chunks/thread.
    auto copy_stage = [&](int s, int k0) {
#pragma unroll
        for (int i = 0; i < 2; ++i) {
            const int idx = tid + i * 256;
            const int r = idx >> 3, c = (idx & 7) * 8;
            async_copy16((unsigned)(uintptr_t)&As[s][r][c],
                         Ab + (size_t)r * K + k0 + c);
        }
#pragma unroll
        for (int i = 0; i < NBCHK; ++i) {
            const int idx = tid + i * 256;
            const int r = idx >> 3, c = (idx & 7) * 8;
            async_copy16((unsigned)(uintptr_t)&Bs[s][r][c],
                         Bb + (size_t)r * K + k0 + c);
        }
    };

    v8f acc[NJ] = {};

    const int nstage = K / KC;
    copy_stage(0, 0);

    for (int s = 0; s < nstage; ++s) {
        const int buf = s & 1;
        if (s + 1 < nstage) {
            copy_stage(buf ^ 1, (s + 1) * KC);
            // own stage-s copies (older, in-order) done when <=NASYNC remain
            asm volatile("s_wait_asynccnt %0" :: "i"(NASYNC) : "memory");
        } else {
            asm volatile("s_wait_asynccnt 0" ::: "memory");
        }
        __syncthreads();   // every wave waited its own copies -> tile visible

        // Preload ALL fragments for this stage, then WMMA back-to-back.
        Frag a[2];
        Frag b[2][NJ];
#pragma unroll
        for (int ks = 0; ks < 2; ++ks) {
            // A frag: lanes 0-15 M=lane (K 0..7|16..23); lanes 16-31 K+8
            const int ar = wr * 16 + (lane & 15);
            const int ac = ks * 32 + ((lane >> 4) << 3);
            a[ks].q[0] = *(const v4u*)&As[buf][ar][ac];
            a[ks].q[1] = *(const v4u*)&As[buf][ar][ac + 16];
#pragma unroll
            for (int j = 0; j < NJ; ++j) {
                // B frag: lanes 0-15 N=lane (K 0..7|16..23); lanes 16-31 K+8
                const int br = wc * (BN_ / 2) + j * 16 + (lane & 15);
                const int bc = ks * 32 + ((lane >> 4) << 3);
                b[ks][j].q[0] = *(const v4u*)&Bs[buf][br][bc];
                b[ks][j].q[1] = *(const v4u*)&Bs[buf][br][bc + 16];
            }
        }
#pragma unroll
        for (int ks = 0; ks < 2; ++ks)
#pragma unroll
            for (int j = 0; j < NJ; ++j)
                acc[j] = wmma_bf16(a[ks], b[ks][j], acc[j]);

        __syncthreads();   // tile consumed; safe to recycle buffer
    }

    // C/D layout: VGPR g -> M = rbase + g; N = ncol0 + 16j + (lane&15)
    const int cl    = lane & 15;
    const int rbase = m0 + wr * 16 + ((lane >> 4) << 3);
    const int ncol0 = n0 + wc * (BN_ / 2);

    float dt = 0.0f;
    if constexpr (EPI == 2) dt = ts[tidx + 1] - ts[tidx];

#pragma unroll
    for (int j = 0; j < NJ; ++j) {
        const int col = ncol0 + j * 16 + cl;
        const float bv = bias[col];
#pragma unroll
        for (int g = 0; g < 8; ++g) {
            const int row    = rbase + g;
            const float gval = acc[j][g] + bv;
            if constexpr (EPI == 0) {
                Dh[(size_t)row * N + col] = to_bf16(fast_tanh(gval));
            } else if constexpr (EPI == 1) {
                Df[(size_t)row * N + col] = gval;
            } else {
                const size_t i  = (size_t)row * N + col;  // N == kH here
                const float  yv = y[i];
                if (mode == 0) {
                    kv1[i] = gval;
                    ubf[i] = to_bf16(yv + dt * gval * (1.0f / 3.0f));
                } else if (mode == 1) {
                    kv2[i] = gval;
                    ubf[i] = to_bf16(yv + dt * (gval - kv1[i] * (1.0f / 3.0f)));
                } else if (mode == 2) {
                    kv3[i] = gval;
                    ubf[i] = to_bf16(yv + dt * (kv1[i] - kv2[i] + gval));
                } else {
                    const float yn =
                        yv + dt * (kv1[i] + 3.0f * (kv2[i] + kv3[i]) + gval) * 0.125f;
                    y[i]   = yn;
                    ybf[i] = to_bf16(yn);
                }
            }
        }
    }
}

// ---------------------------------------------------------------------------
// Transpose + f32 -> bf16 convert: src[R,C] f32 -> dst[C,R] bf16 (LDS tiled)
// ---------------------------------------------------------------------------
__global__ __launch_bounds__(256)
void transpose_bf16_kernel(const float* __restrict__ src, __bf16* __restrict__ dst,
                           int R, int C)
{
    __shared__ float tile[32][33];
    const int c0 = blockIdx.x * 32;
    const int r0 = blockIdx.y * 32;
    const int tx = threadIdx.x & 31;
    const int ty = threadIdx.x >> 5;
#pragma unroll
    for (int i = 0; i < 4; ++i) {
        const int a = ty + i * 8;
        tile[a][tx] = src[(size_t)(r0 + a) * C + (c0 + tx)];
    }
    __syncthreads();
#pragma unroll
    for (int i = 0; i < 4; ++i) {
        const int s = ty + i * 8;
        dst[(size_t)(c0 + s) * R + (r0 + tx)] = to_bf16(tile[tx][s]);
    }
}

// ---------------------------------------------------------------------------
// State init: y = z (f32), ybf = bf16(z)
// ---------------------------------------------------------------------------
__global__ __launch_bounds__(256)
void init_state_kernel(const float* __restrict__ z, float* __restrict__ y,
                       __bf16* __restrict__ ybf, int n)
{
    const int i = blockIdx.x * 256 + threadIdx.x;
    if (i < n) {
        const float v = z[i];
        y[i]   = v;
        ybf[i] = to_bf16(v);
    }
}

// ---------------------------------------------------------------------------
// Softmax over C=512 per row; one wave per row, 16 elems per lane.
// ---------------------------------------------------------------------------
__device__ __forceinline__ float max4(v4f a) {
    return fmaxf(fmaxf(a.x, a.y), fmaxf(a.z, a.w));
}
__device__ __forceinline__ float sum4(v4f a) { return (a.x + a.y) + (a.z + a.w); }

__global__ __launch_bounds__(256)
void softmax512_kernel(const float* __restrict__ logits, float* __restrict__ out)
{
    const int wave = threadIdx.x >> 5;
    const int lane = threadIdx.x & 31;
    const int row  = blockIdx.x * 8 + wave;
    const float* src = logits + (size_t)row * 512 + lane * 16;
    v4f v[4];
#pragma unroll
    for (int i = 0; i < 4; ++i) v[i] = *(const v4f*)(src + 4 * i);

    float m = fmaxf(fmaxf(max4(v[0]), max4(v[1])), fmaxf(max4(v[2]), max4(v[3])));
#pragma unroll
    for (int off = 16; off > 0; off >>= 1) m = fmaxf(m, __shfl_xor(m, off));

    float s = 0.0f;
#pragma unroll
    for (int i = 0; i < 4; ++i) {
        v[i].x = __expf(v[i].x - m);
        v[i].y = __expf(v[i].y - m);
        v[i].z = __expf(v[i].z - m);
        v[i].w = __expf(v[i].w - m);
        s += sum4(v[i]);
    }
#pragma unroll
    for (int off = 16; off > 0; off >>= 1) s += __shfl_xor(s, off);

    const float r = 1.0f / s;
    float* dst = out + (size_t)row * 512 + lane * 16;
#pragma unroll
    for (int i = 0; i < 4; ++i) {
        v4f o = v[i];
        o.x *= r; o.y *= r; o.z *= r; o.w *= r;
        *(v4f*)(dst + 4 * i) = o;
    }
}

// ---------------------------------------------------------------------------
// Host orchestration
// ---------------------------------------------------------------------------
extern "C" void kernel_launch(void* const* d_in, const int* in_sizes, int n_in,
                              void* d_out, int out_size, void* d_ws, size_t ws_size,
                              hipStream_t stream)
{
    (void)in_sizes; (void)n_in; (void)out_size; (void)ws_size;

    const float* z  = (const float*)d_in[0];   // [B,H]
    const float* ts = (const float*)d_in[1];   // [T]
    const float* W1 = (const float*)d_in[2];   // [H,OH]
    const float* b1 = (const float*)d_in[3];   // [OH]
    const float* W2 = (const float*)d_in[4];   // [OH,H]
    const float* b2 = (const float*)d_in[5];   // [H]
    const float* Wf = (const float*)d_in[6];   // [H,C]
    const float* bf = (const float*)d_in[7];   // [C]
    float* out = (float*)d_out;                // [T,B,C]

    // ---- workspace carve-up (256B aligned) ----
    size_t off = 0;
    auto carve = [&](size_t bytes) -> char* {
        char* p = (char*)d_ws + off;
        off += (bytes + 255) & ~(size_t)255;
        return p;
    };
    __bf16* W1t  = (__bf16*)carve((size_t)kOH * kH * 2);  // [OH,H]
    __bf16* W2t  = (__bf16*)carve((size_t)kH * kOH * 2);  // [H,OH]
    __bf16* Wft  = (__bf16*)carve((size_t)kC * kH * 2);   // [C,H]
    __bf16* ybf  = (__bf16*)carve((size_t)kB * kH * 2);
    __bf16* ubf  = (__bf16*)carve((size_t)kB * kH * 2);
    __bf16* hbf  = (__bf16*)carve((size_t)kB * kOH * 2);  // tanh activations
    float*  y    = (float*)carve((size_t)kB * kH * 4);
    float*  kv1  = (float*)carve((size_t)kB * kH * 4);
    float*  kv2  = (float*)carve((size_t)kB * kH * 4);
    float*  kv3  = (float*)carve((size_t)kB * kH * 4);
    float*  lgts = (float*)carve((size_t)kB * kC * 4);

    const dim3 blk(256);
    const int  BH = kB * kH;

    transpose_bf16_kernel<<<dim3(kOH / 32, kH / 32), blk, 0, stream>>>(W1, W1t, kH, kOH);
    transpose_bf16_kernel<<<dim3(kH / 32, kOH / 32), blk, 0, stream>>>(W2, W2t, kOH, kH);
    transpose_bf16_kernel<<<dim3(kC / 32, kH / 32),  blk, 0, stream>>>(Wf, Wft, kH, kC);

    init_state_kernel<<<dim3(BH / 256), blk, 0, stream>>>(z, y, ybf, BH);

    const dim3 g1(kOH / 128, kB / BM);   // GEMM1: 32x8 blocks
    const dim3 g2(kH  / 64,  kB / BM);   // GEMM2: 16x8 blocks
    const dim3 gp(kC  / 64,  kB / BM);   // proj :  8x8 blocks
    const dim3 gs(kB / 8);

    for (int t = 0; t < kT; ++t) {
        // project current state -> softmax -> out[t]
        wmma_gemm_kernel<1, 64><<<gp, blk, 0, stream>>>(
            ybf, Wft, bf, lgts, nullptr, kB, kC, kH,
            0, nullptr, 0, nullptr, nullptr, nullptr, nullptr, nullptr, nullptr);
        softmax512_kernel<<<gs, blk, 0, stream>>>(lgts, out + (size_t)t * kB * kC);

        if (t == kT - 1) break;

        // RK4 (3/8 rule): 4 evaluations of f; combine fused into GEMM2 epilogue
        const __bf16* uin = ybf;
        for (int mode = 0; mode < 4; ++mode) {
            wmma_gemm_kernel<0, 128><<<g1, blk, 0, stream>>>(
                uin, W1t, b1, nullptr, hbf, kB, kOH, kH,
                0, nullptr, 0, nullptr, nullptr, nullptr, nullptr, nullptr, nullptr);
            wmma_gemm_kernel<2, 64><<<g2, blk, 0, stream>>>(
                hbf, W2t, b2, nullptr, nullptr, kB, kH, kOH,
                mode, ts, t, y, kv1, kv2, kv3, ubf, ybf);
            uin = ubf;
        }
    }
}
